// Pointnet2MSG_18691697672678
// MI455X (gfx1250) — matrix operand order repaired and checked by reading, not compiled, not run on hardware
//
#include <hip/hip_runtime.h>

// ---------------------------------------------------------------------------
// PointNet++ MSG forward for MI455X (gfx1250, wave32, WMMA).
// All activations are bf16, stored point-major [pt][channel] with channels
// padded to a multiple of 32; weights pre-converted to bf16 [Opad32][Kpad32].
// GEMM: 2x2 register-blocked WMMA (32x32 output per wave, 4 WMMAs / K-step).
// ---------------------------------------------------------------------------

typedef __bf16 bf;
typedef __attribute__((ext_vector_type(16))) __bf16 v16bf;
typedef __attribute__((ext_vector_type(8)))  __bf16 v8bf;
typedef __attribute__((ext_vector_type(8)))  float  v8f;

static __host__ __device__ inline int ru32(int x) { return (x + 31) & ~31; }

static __device__ inline v16bf cat8(v8bf lo, v8bf hi) {
    return __builtin_shufflevector(lo, hi,
        0, 1, 2, 3, 4, 5, 6, 7, 8, 9, 10, 11, 12, 13, 14, 15);
}

// ------------------------------ GEMM (WMMA) --------------------------------
// D[n, m] = (relu?) sum_k A[m, k] * B[n, k]
//   A: bf16 [O][Kp]   (O % 32 == 0, Kp % 32 == 0, zero padded)
//   B: bf16 [N][Kp]   (N % 32 == 0)
//   C: bf16 [N][ldc]  (ldc % 32 == 0)
// Each wave produces a 32(M) x 32(N) block: 4 accumulators, 2 A + 2 B frags.
__global__ void gemm_wmma_kernel(const bf* __restrict__ A,
                                 const bf* __restrict__ B,
                                 bf* __restrict__ C,
                                 int O, int Kp, int N, int ldc, int relu) {
    const int wave = blockIdx.x * (blockDim.x >> 5) + ((int)threadIdx.x >> 5);
    const int tilesN = N >> 5;
    const int tilesO = O >> 5;
    if (wave >= tilesN * tilesO) return;       // wave-uniform
    const int to = wave / tilesN;
    const int tn = wave - to * tilesN;
    const int lane = threadIdx.x & 31;
    const int half = lane >> 4;                // 0|1
    const int l    = lane & 15;

    const bf* Ap0 = A + (size_t)(to * 32 + l) * Kp + half * 16;  // 32B aligned
    const bf* Ap1 = Ap0 + (size_t)16 * Kp;
    const bf* Bp0 = B + (size_t)(tn * 32 + l) * Kp + half * 16;  // 32B aligned
    const bf* Bp1 = Bp0 + (size_t)16 * Kp;

    v8f acc00 = {0.f, 0.f, 0.f, 0.f, 0.f, 0.f, 0.f, 0.f};
    v8f acc01 = acc00, acc10 = acc00, acc11 = acc00;

    for (int k0 = 0; k0 < Kp; k0 += 32) {
        __builtin_prefetch(Bp0 + 64, 0, 0);
        __builtin_prefetch(Bp1 + 64, 0, 0);
        v16bf a0 = cat8(*(const v8bf*)(Ap0), *(const v8bf*)(Ap0 + 8));
        v16bf a1 = cat8(*(const v8bf*)(Ap1), *(const v8bf*)(Ap1 + 8));
        v16bf b0 = *(const v16bf*)(Bp0);
        v16bf b1 = *(const v16bf*)(Bp1);
        acc00 = __builtin_amdgcn_wmma_f32_16x16x32_bf16(
            false, a0, false, b0, (short)0, acc00, false, false);
        acc01 = __builtin_amdgcn_wmma_f32_16x16x32_bf16(
            false, a0, false, b1, (short)0, acc01, false, false);
        acc10 = __builtin_amdgcn_wmma_f32_16x16x32_bf16(
            false, a1, false, b0, (short)0, acc10, false, false);
        acc11 = __builtin_amdgcn_wmma_f32_16x16x32_bf16(
            false, a1, false, b1, (short)0, acc11, false, false);
        Ap0 += 32; Ap1 += 32; Bp0 += 32; Bp1 += 32;
    }

    // C store: acc[mi][ni] lane -> row n = tn*32+ni*16+l,
    //          8 contiguous channels at m = to*32+mi*16+half*8
    v8f accs[2][2] = {{acc00, acc01}, {acc10, acc11}};
#pragma unroll
    for (int mi = 0; mi < 2; ++mi) {
#pragma unroll
        for (int ni = 0; ni < 2; ++ni) {
            v8bf cv;
#pragma unroll
            for (int r = 0; r < 8; ++r) {
                float v = accs[mi][ni][r];
                if (relu) v = fmaxf(v, 0.f);
                cv[r] = (bf)v;
            }
            *(v8bf*)(C + (size_t)(tn * 32 + ni * 16 + l) * ldc
                       + (size_t)(to * 32 + mi * 16 + half * 8)) = cv;
        }
    }
}

// --------------------- weight f32 -> padded bf16 ---------------------------
__global__ void convert_w_kernel(const float* __restrict__ src, bf* __restrict__ dst,
                                 int O, int K, int Op, int Kp) {
    long i = (long)blockIdx.x * blockDim.x + threadIdx.x;
    if (i >= (long)Op * Kp) return;
    int o = (int)(i / Kp), k = (int)(i % Kp);
    float v = (o < O && k < K) ? src[(size_t)o * K + k] : 0.f;
    dst[i] = (bf)v;
}

// ------------------------- Farthest point sampling -------------------------
__global__ void fps_kernel(const float* __restrict__ xyz, int N, int npoint,
                           int* __restrict__ out_idx, float* __restrict__ dist) {
    __shared__ float s_val[1024];
    __shared__ int   s_idx[1024];
    __shared__ int   s_cur;
    __shared__ float s_cx, s_cy, s_cz;
    const int tid = threadIdx.x;
    for (int i = tid; i < N; i += blockDim.x) dist[i] = 1e10f;
    if (tid == 0) { s_cur = 0; out_idx[0] = 0; }
    __syncthreads();
    for (int it = 0; it < npoint; ++it) {
        if (tid == 0) {
            int c = s_cur;
            s_cx = xyz[c * 3 + 0]; s_cy = xyz[c * 3 + 1]; s_cz = xyz[c * 3 + 2];
        }
        __syncthreads();
        float best = -1.f; int bi = 0;
        for (int i = tid; i < N; i += blockDim.x) {
            float dx = xyz[i * 3 + 0] - s_cx;
            float dy = xyz[i * 3 + 1] - s_cy;
            float dz = xyz[i * 3 + 2] - s_cz;
            float d = dx * dx + dy * dy + dz * dz;
            float dm = fminf(dist[i], d);
            dist[i] = dm;
            if (dm > best) { best = dm; bi = i; }
        }
        s_val[tid] = best; s_idx[tid] = bi;
        __syncthreads();
        for (int s = blockDim.x >> 1; s > 0; s >>= 1) {
            if (tid < s && s_val[tid + s] > s_val[tid]) {
                s_val[tid] = s_val[tid + s]; s_idx[tid] = s_idx[tid + s];
            }
            __syncthreads();
        }
        if (tid == 0) {
            s_cur = s_idx[0];
            if (it + 1 < npoint) out_idx[it + 1] = s_idx[0];
        }
        __syncthreads();
    }
}

__global__ void gather_xyz_kernel(const float* __restrict__ xyz,
                                  const int* __restrict__ idx,
                                  float* __restrict__ out, int S) {
    int s = blockIdx.x * blockDim.x + threadIdx.x;
    if (s >= S) return;
    int id = idx[s];
    out[s * 3 + 0] = xyz[id * 3 + 0];
    out[s * 3 + 1] = xyz[id * 3 + 1];
    out[s * 3 + 2] = xyz[id * 3 + 2];
}

// ------------------------------- Ball query --------------------------------
__global__ void ball_query_kernel(const float* __restrict__ xyz,
                                  const float* __restrict__ nxyz,
                                  int* __restrict__ idx,
                                  int N, int S, float r2, int ns) {
    int s = blockIdx.x * blockDim.x + threadIdx.x;
    if (s >= S) return;
    float cx = nxyz[s * 3 + 0], cy = nxyz[s * 3 + 1], cz = nxyz[s * 3 + 2];
    int cnt = 0;
    for (int i = 0; i < N && cnt < ns; ++i) {
        float dx = xyz[i * 3 + 0] - cx;
        float dy = xyz[i * 3 + 1] - cy;
        float dz = xyz[i * 3 + 2] - cz;
        if (dx * dx + dy * dy + dz * dz < r2) idx[(size_t)s * ns + cnt++] = i;
    }
    int fill = (cnt > 0) ? idx[(size_t)s * ns] : 0;
    for (; cnt < ns; ++cnt) idx[(size_t)s * ns + cnt] = fill;
}

// ----- grouping: X[(s*ns+i)][c] = {xyz-diff (3), feats (Cf), zeros (pad)} ---
__global__ void group_kernel(const float* __restrict__ xyz,
                             const float* __restrict__ nxyz,
                             const bf* __restrict__ feats,
                             const int* __restrict__ idx,
                             bf* __restrict__ out,
                             int S, int ns, int Cf, int ldf, int Kp) {
    long i = (long)blockIdx.x * blockDim.x + threadIdx.x;
    long cols = (long)S * ns;
    if (i >= cols) return;
    int s = (int)(i / ns);
    int id = idx[i];
    bf* o = out + (size_t)i * Kp;
    o[0] = (bf)(xyz[id * 3 + 0] - nxyz[s * 3 + 0]);
    o[1] = (bf)(xyz[id * 3 + 1] - nxyz[s * 3 + 1]);
    o[2] = (bf)(xyz[id * 3 + 2] - nxyz[s * 3 + 2]);
    const bf* f = feats + (size_t)id * ldf;
    for (int c = 0; c < Cf; ++c) o[3 + c] = f[c];
    for (int c = 3 + Cf; c < Kp; ++c) o[c] = (bf)0.f;
}

// max over ns samples: out[s][chOff+c] = max_n in[(s*ns+n)][c]
__global__ void maxpool_kernel(const bf* __restrict__ in, bf* __restrict__ out,
                               int C, int S, int ns, int ldin, int ldout, int chOff) {
    long i = (long)blockIdx.x * blockDim.x + threadIdx.x;
    if (i >= (long)C * S) return;
    int s = (int)(i / C), c = (int)(i % C);
    const bf* p = in + (size_t)s * ns * ldin + c;
    float mx = (float)p[0];
    for (int n = 1; n < ns; ++n) mx = fmaxf(mx, (float)p[(size_t)n * ldin]);
    out[(size_t)s * ldout + chOff + c] = (bf)mx;
}

// ------------------- 3-NN inverse-distance interpolation -------------------
__global__ void three_interp_kernel(const float* __restrict__ unk,
                                    const float* __restrict__ kn,
                                    const bf* __restrict__ kf,
                                    bf* __restrict__ out,
                                    int n, int m, int C, int ldk, int ldo) {
    int i = blockIdx.x * blockDim.x + threadIdx.x;
    if (i >= n) return;
    float ux = unk[i * 3 + 0], uy = unk[i * 3 + 1], uz = unk[i * 3 + 2];
    float d0 = 1e30f, d1 = 1e30f, d2 = 1e30f;
    int i0 = 0, i1 = 0, i2 = 0;
    for (int j = 0; j < m; ++j) {
        float dx = kn[j * 3 + 0] - ux;
        float dy = kn[j * 3 + 1] - uy;
        float dz = kn[j * 3 + 2] - uz;
        float d = dx * dx + dy * dy + dz * dz;
        if (d < d0)      { d2 = d1; i2 = i1; d1 = d0; i1 = i0; d0 = d; i0 = j; }
        else if (d < d1) { d2 = d1; i2 = i1; d1 = d;  i1 = j; }
        else if (d < d2) { d2 = d;  i2 = j; }
    }
    float w0 = 1.f / (d0 + 1e-8f), w1 = 1.f / (d1 + 1e-8f), w2 = 1.f / (d2 + 1e-8f);
    float ws = w0 + w1 + w2;
    w0 /= ws; w1 /= ws; w2 /= ws;
    const bf* r0 = kf + (size_t)i0 * ldk;
    const bf* r1 = kf + (size_t)i1 * ldk;
    const bf* r2 = kf + (size_t)i2 * ldk;
    bf* o = out + (size_t)i * ldo;
    for (int c = 0; c < C; ++c)
        o[c] = (bf)((float)r0[c] * w0 + (float)r1[c] * w1 + (float)r2[c] * w2);
}

// generic channel-block copy: dst[p][c] = src[p][c]
__global__ void copy_chan_kernel(const bf* __restrict__ src, bf* __restrict__ dst,
                                 int npts, int nch, int srcLd, int dstLd) {
    long i = (long)blockIdx.x * blockDim.x + threadIdx.x;
    if (i >= (long)npts * nch) return;
    int p = (int)(i / nch), c = (int)(i % nch);
    dst[(size_t)p * dstLd + c] = src[(size_t)p * srcLd + c];
}

// src [R][Cc] -> dst [Cc][R]
__global__ void transpose_bf_kernel(const bf* __restrict__ src, bf* __restrict__ dst,
                                    int R, int Cc) {
    long i = (long)blockIdx.x * blockDim.x + threadIdx.x;
    if (i >= (long)R * Cc) return;
    int r = (int)(i / Cc), c = (int)(i % Cc);
    dst[(size_t)c * R + r] = src[i];
}

// softmax over rows x (stride Cc) for each column y; mat stored [x][y]
__global__ void softmax_col_kernel(bf* __restrict__ mat, int R, int Cc) {
    int y = blockIdx.x * blockDim.x + threadIdx.x;
    if (y >= Cc) return;
    float mx = -1e30f;
    for (int x = 0; x < R; ++x) mx = fmaxf(mx, (float)mat[(size_t)x * Cc + y]);
    float s = 0.f;
    for (int x = 0; x < R; ++x) {
        float e = __expf((float)mat[(size_t)x * Cc + y] - mx);
        mat[(size_t)x * Cc + y] = (bf)e;
        s += e;
    }
    float inv = 1.f / s;
    for (int x = 0; x < R; ++x)
        mat[(size_t)x * Cc + y] = (bf)((float)mat[(size_t)x * Cc + y] * inv);
}

__global__ void relu_add_bf_kernel(const bf* __restrict__ a, bf* __restrict__ b,
                                   long n) {
    long i = (long)blockIdx.x * blockDim.x + threadIdx.x;
    if (i >= n) return;
    b[i] = (bf)((float)b[i] + fmaxf((float)a[i], 0.f));
}

// final feature output: dst[c][colOff + p] = (float) src[p][c]
__global__ void out_feats_kernel(const bf* __restrict__ src, float* __restrict__ dst,
                                 int npts, int C, int ldsrc, int dstLd, int colOff) {
    long i = (long)blockIdx.x * blockDim.x + threadIdx.x;
    if (i >= (long)npts * C) return;
    int p = (int)(i / C), c = (int)(i % C);
    dst[(size_t)c * dstLd + colOff + p] = (float)src[(size_t)p * ldsrc + c];
}

// ---------------------------------------------------------------------------
// Host-side orchestration
// ---------------------------------------------------------------------------
static inline int gridFor(long n, int b) { return (int)((n + b - 1) / b); }

static void launch_gemm(hipStream_t st, const bf* A, const bf* B, bf* C,
                        int O, int Kp, int N, int ldc, int relu) {
    long tiles = (long)(O >> 5) * (N >> 5);    // 32x32 blocks per wave
    const int wavesPerBlk = 8;                 // 256 threads
    gemm_wmma_kernel<<<gridFor(tiles, wavesPerBlk), wavesPerBlk * 32, 0, st>>>(
        A, B, C, O, Kp, N, ldc, relu);
}

static const int NPOINTS_[4] = {4096, 1024, 256, 64};
static const int NS_[2] = {16, 32};
static const float R2_[4][2] = {{0.01f, 0.25f}, {0.25f, 1.f}, {1.f, 4.f}, {4.f, 16.f}};
static const int SA_CHAIN[4][2][4] = {
    {{3, 16, 16, 32},      {3, 32, 32, 64}},
    {{99, 64, 64, 128},    {99, 64, 96, 128}},
    {{259, 128, 196, 256}, {259, 128, 196, 256}},
    {{515, 256, 256, 512}, {515, 256, 384, 512}}};
static const int COUT_[4] = {96, 256, 512, 1024};
static const int FPCH[4][3] = {{256, 128, 128}, {608, 256, 256},
                               {768, 512, 512}, {1536, 512, 512}};
static const int NPT = 16384;

struct Frame {
    const float* xyz0;
    float *lx1, *lx2, *lx3, *lx4;
    bf *lf1, *lf2, *lf3, *lf4, *lf0;
};

extern "C" void kernel_launch(void* const* d_in, const int* in_sizes, int n_in,
                              void* d_out, int out_size, void* d_ws, size_t ws_size,
                              hipStream_t stream) {
    (void)in_sizes; (void)n_in; (void)out_size; (void)ws_size;
    const float* pc = (const float*)d_in[0];

    // ------- deterministic bump allocation over d_ws -------
    char* base = (char*)d_ws;
    size_t off = 0;
    auto af  = [&](size_t n) { float* p = (float*)(base + off);
                               off += ((n * 4 + 255) & ~(size_t)255); return p; };
    auto ab  = [&](size_t n) { bf* p = (bf*)(base + off);
                               off += ((n * 2 + 255) & ~(size_t)255); return p; };
    auto ain = [&](size_t n) { int* p = (int*)(base + off);
                               off += ((n * 4 + 255) & ~(size_t)255); return p; };

    float* dist    = af(NPT);
    int*   fps_idx = ain(4096);
    int*   bq_idx  = ain(4096 * 32);
    bf*    ping    = ab((size_t)131072 * 64);
    bf*    pong    = ab((size_t)131072 * 64);
    bf*    nlflat  = ab((size_t)2048 * 512);
    bf*    nltmp   = ab((size_t)2048 * 512);
    bf*    thb     = ab((size_t)2048 * 256);
    bf*    phb     = ab((size_t)2048 * 256);
    bf*    gbb     = ab((size_t)2048 * 256);
    bf*    gbT     = ab((size_t)256 * 2048);
    bf*    ybb     = ab((size_t)2048 * 256);
    bf*    corr    = ab((size_t)2048 * 2048);

    // -------- convert all weights to padded bf16 once --------
    auto convW = [&](const float* src, int O, int K) -> bf* {
        int Op = ru32(O), Kp = ru32(K);
        bf* dst = ab((size_t)Op * Kp);
        convert_w_kernel<<<gridFor((long)Op * Kp, 256), 256, 0, stream>>>(
            src, dst, O, K, Op, Kp);
        return dst;
    };
    bf* saWc[4][2][3];
    for (int k = 0; k < 4; ++k)
        for (int j = 0; j < 2; ++j)
            for (int t = 0; t < 3; ++t)
                saWc[k][j][t] = convW((const float*)d_in[1 + (k * 2 + j) * 3 + t],
                                      SA_CHAIN[k][j][t + 1], SA_CHAIN[k][j][t]);
    bf* fpWc[4][2];
    for (int k = 0; k < 4; ++k)
        for (int t = 0; t < 2; ++t)
            fpWc[k][t] = convW((const float*)d_in[25 + k * 2 + t],
                               FPCH[k][t + 1], FPCH[k][t]);
    bf* WtC = convW((const float*)d_in[33], 256, 512);
    bf* WpC = convW((const float*)d_in[34], 256, 512);
    bf* WgC = convW((const float*)d_in[35], 256, 512);
    bf* WrC = convW((const float*)d_in[36], 512, 256);

    Frame fr[2];
    for (int f = 0; f < 2; ++f) {
        fr[f].xyz0 = pc + (f == 1 ? 0 : (size_t)NPT * 3);   // frame1 = first half
        fr[f].lx1 = af(4096 * 3); fr[f].lx2 = af(1024 * 3);
        fr[f].lx3 = af(256 * 3);  fr[f].lx4 = af(64 * 3);
        fr[f].lf1 = ab((size_t)4096 * 256);   // ld 96 (SA) then 256 (FP)
        fr[f].lf2 = ab((size_t)1024 * 512);   // ld 256 (SA) then 512 (FP)
        fr[f].lf3 = ab((size_t)256 * 512);
        fr[f].lf4 = ab((size_t)64 * 1024);
        fr[f].lf0 = ab((size_t)NPT * 128);
    }

    // ============================ SA levels ============================
    for (int f = 0; f < 2; ++f) {
        Frame& F = fr[f];
        const float* cx = F.xyz0; int Nin = NPT;
        const bf* cf = nullptr; int Cin = 0, ldf = 0;
        float* lxs[5] = {nullptr, F.lx1, F.lx2, F.lx3, F.lx4};
        bf*    lfs[5] = {nullptr, F.lf1, F.lf2, F.lf3, F.lf4};
        for (int k = 0; k < 4; ++k) {
            int S = NPOINTS_[k];
            fps_kernel<<<1, 1024, 0, stream>>>(cx, Nin, S, fps_idx, dist);
            gather_xyz_kernel<<<gridFor(S, 256), 256, 0, stream>>>(cx, fps_idx,
                                                                   lxs[k + 1], S);
            int chOff = 0;
            for (int j = 0; j < 2; ++j) {
                int ns = NS_[j];
                long cols = (long)S * ns;
                const int* ch = SA_CHAIN[k][j];
                int KpIn = ru32(ch[0]);
                ball_query_kernel<<<gridFor(S, 256), 256, 0, stream>>>(
                    cx, lxs[k + 1], bq_idx, Nin, S, R2_[k][j], ns);
                group_kernel<<<gridFor(cols, 256), 256, 0, stream>>>(
                    cx, lxs[k + 1], cf, bq_idx, ping, S, ns, Cin, ldf, KpIn);
                bf* cur = ping; bf* nxt = pong;
                int Kp = KpIn;
                for (int t = 0; t < 3; ++t) {
                    int Op = ru32(ch[t + 1]);
                    launch_gemm(stream, saWc[k][j][t], cur, nxt, Op, Kp,
                                (int)cols, Op, 1);
                    bf* tmp = cur; cur = nxt; nxt = tmp;
                    Kp = Op;
                }
                maxpool_kernel<<<gridFor((long)ch[3] * S, 256), 256, 0, stream>>>(
                    cur, lfs[k + 1], ch[3], S, ns, ru32(ch[3]), COUT_[k], chOff);
                chOff += ch[3];
            }
            cx = lxs[k + 1]; cf = lfs[k + 1]; Cin = COUT_[k]; ldf = COUT_[k]; Nin = S;
        }
    }

    // ============================ FP 3, FP 2 ============================
    auto run_fp = [&](const float* unk, int n, const float* kn, int m,
                      const bf* skip, int Cs, int skipLd,
                      const bf* kf, int Ck, int kfLd,
                      bf* W0, bf* W1, const int* ch, bf* outf) {
        // X = [interp(Ck) | skip(Cs)]  in ping, ld = ch[0] (multiple of 32)
        three_interp_kernel<<<gridFor(n, 128), 128, 0, stream>>>(
            unk, kn, kf, ping, n, m, Ck, kfLd, ch[0]);
        if (skip && Cs > 0)
            copy_chan_kernel<<<gridFor((long)n * Cs, 256), 256, 0, stream>>>(
                skip, ping + Ck, n, Cs, skipLd, ch[0]);
        launch_gemm(stream, W0, ping, pong, ru32(ch[1]), ch[0], n, ru32(ch[1]), 1);
        launch_gemm(stream, W1, pong, outf, ru32(ch[2]), ru32(ch[1]), n, ru32(ch[2]), 1);
    };
    for (int f = 0; f < 2; ++f) {
        Frame& F = fr[f];
        run_fp(F.lx3, 256, F.lx4, 64, F.lf3, 512, 512, F.lf4, 1024, 1024,
               fpWc[3][0], fpWc[3][1], FPCH[3], F.lf3);
        run_fp(F.lx2, 1024, F.lx3, 256, F.lf2, 256, 256, F.lf3, 512, 512,
               fpWc[2][0], fpWc[2][1], FPCH[2], F.lf2);
    }

    // ============================ Non-local =============================
    // nlflat[x][c], x = t*1024 + pt, t=0 -> frame0
    for (int f = 0; f < 2; ++f)
        copy_chan_kernel<<<gridFor((long)1024 * 512, 256), 256, 0, stream>>>(
            fr[f].lf2, nlflat + (size_t)f * 1024 * 512, 1024, 512, 512, 512);
    launch_gemm(stream, WtC, nlflat, thb, 256, 512, 2048, 256, 0);
    launch_gemm(stream, WpC, nlflat, phb, 256, 512, 2048, 256, 0);
    launch_gemm(stream, WgC, nlflat, gbb, 256, 512, 2048, 256, 1);
    // corr[x][y] = sum_c th[x][c] * ph[y][c]   (A = ph rows=y, B = th rows=x)
    launch_gemm(stream, phb, thb, corr, 2048, 256, 2048, 2048, 0);
    softmax_col_kernel<<<gridFor(2048, 256), 256, 0, stream>>>(corr, 2048, 2048);
    // y[x][c] = sum_y g[y][c] * corr[x][y]     (A = g^T [c][y], B = corr [x][y])
    transpose_bf_kernel<<<gridFor((long)2048 * 256, 256), 256, 0, stream>>>(
        gbb, gbT, 2048, 256);
    launch_gemm(stream, gbT, corr, ybb, 256, 2048, 2048, 256, 0);
    launch_gemm(stream, WrC, ybb, nltmp, 512, 256, 2048, 512, 0);
    relu_add_bf_kernel<<<gridFor((long)2048 * 512, 256), 256, 0, stream>>>(
        nltmp, nlflat, (long)2048 * 512);
    for (int f = 0; f < 2; ++f)
        copy_chan_kernel<<<gridFor((long)1024 * 512, 256), 256, 0, stream>>>(
            nlflat + (size_t)f * 1024 * 512, fr[f].lf2, 1024, 512, 512, 512);

    // ============================ FP 1, FP 0 ============================
    for (int f = 0; f < 2; ++f) {
        Frame& F = fr[f];
        run_fp(F.lx1, 4096, F.lx2, 1024, F.lf1, 96, 96, F.lf2, 512, 512,
               fpWc[1][0], fpWc[1][1], FPCH[1], F.lf1);
        run_fp(F.xyz0, NPT, F.lx1, 4096, nullptr, 0, 0, F.lf1, 256, 256,
               fpWc[0][0], fpWc[0][1], FPCH[0], F.lf0);
    }

    // ============================ Outputs ===============================
    float* out = (float*)d_out;
    hipMemcpyAsync(out, pc, (size_t)2 * NPT * 3 * sizeof(float),
                   hipMemcpyDeviceToDevice, stream);
    float* featOut = out + (size_t)2 * NPT * 3;        // [128][2*NPT] f32
    out_feats_kernel<<<gridFor((long)NPT * 128, 256), 256, 0, stream>>>(
        fr[1].lf0, featOut, NPT, 128, 128, 2 * NPT, 0);
    out_feats_kernel<<<gridFor((long)NPT * 128, 256), 256, 0, stream>>>(
        fr[0].lf0, featOut, NPT, 128, 128, 2 * NPT, NPT);
}
